// AttentionModel_48017734369632
// MI455X (gfx1250) — matrix operand order repaired and checked by reading, not compiled, not run on hardware
//
#include <hip/hip_runtime.h>
#include <math.h>

#define B_  512
#define F_  8
#define S_  256
#define H_  512
#define T_  64
#define EPS_ 1e-5f

typedef __attribute__((ext_vector_type(16))) __bf16 v16bf;
typedef __attribute__((ext_vector_type(8)))  float  v8f;
typedef __attribute__((ext_vector_type(4)))  unsigned int u32x4;

union Frag { u32x4 u[2]; v16bf v; };

__device__ __forceinline__ unsigned short f2bf(float f) {
    union { float f; unsigned u; } x; x.f = f;
    unsigned u = x.u;
    u += 0x7FFFu + ((u >> 16) & 1u);          // round-to-nearest-even
    return (unsigned short)(u >> 16);
}
__device__ __forceinline__ float bf2f(unsigned short s) {
    union { unsigned u; float f; } x; x.u = ((unsigned)s) << 16;
    return x.f;
}
__device__ __forceinline__ float sigf(float x) { return 1.f / (1.f + expf(-x)); }

// ---------------------------------------------------------------------------
// Encoder BatchNorm stats: per-channel (F=8) mean/var over B*S, fused to
// scale/shift so normalization is a single FMA later.
// ---------------------------------------------------------------------------
__global__ __launch_bounds__(256) void bn_e_stats_k(const float* __restrict__ x1,
        const float* __restrict__ g, const float* __restrict__ bb,
        float* __restrict__ scaleE, float* __restrict__ shiftE) {
    __shared__ float s1[256], s2[256];
    const int f = blockIdx.x;
    float sum = 0.f, sq = 0.f;
    for (int i = threadIdx.x; i < B_ * S_; i += 256) {
        int b = i >> 8, s = i & 255;
        float v = x1[(size_t)b * (F_ * S_) + (size_t)f * S_ + s];
        sum += v; sq += v * v;
    }
    s1[threadIdx.x] = sum; s2[threadIdx.x] = sq; __syncthreads();
    for (int st = 128; st > 0; st >>= 1) {
        if (threadIdx.x < st) { s1[threadIdx.x] += s1[threadIdx.x + st]; s2[threadIdx.x] += s2[threadIdx.x + st]; }
        __syncthreads();
    }
    if (threadIdx.x == 0) {
        const float inv = 1.f / (float)(B_ * S_);
        float m = s1[0] * inv;
        float var = s2[0] * inv - m * m;
        float sc = g[f] * rsqrtf(var + EPS_);
        scaleE[f] = sc; shiftE[f] = bb[f] - m * sc;
    }
}

// xn[t][b][f] = normalized x1 (so the encoder GEMM scalar tail reads 8
// contiguous floats per row).
__global__ __launch_bounds__(256) void build_xn_k(const float* __restrict__ x1,
        const float* __restrict__ scaleE, const float* __restrict__ shiftE,
        float* __restrict__ xn) {
    int idx = blockIdx.x * 256 + threadIdx.x;      // S*B*F total
    int t = idx / (B_ * F_);
    int r = idx % (B_ * F_);
    int b = r >> 3, f = r & 7;
    xn[idx] = x1[(size_t)b * (F_ * S_) + (size_t)f * S_ + t] * scaleE[f] + shiftE[f];
}

// Generic fp32 -> bf16 weight conversion with row stride / column offset
// (lets us slice the first 2 columns off attn_W / comb_W).
__global__ __launch_bounds__(256) void cvt_bf16_k(const float* __restrict__ src,
        unsigned short* __restrict__ dst, int cols, int srcStride, int srcOff, int total) {
    int idx = blockIdx.x * 256 + threadIdx.x;
    if (idx >= total) return;
    int r = idx / cols, c = idx % cols;
    dst[idx] = f2bf(src[(size_t)r * srcStride + srcOff + c]);
}

__global__ __launch_bounds__(256) void vec_add_k(const float* a, const float* b, float* o, int n) {
    int i = blockIdx.x * 256 + threadIdx.x;
    if (i < n) o[i] = a[i] + b[i];
}

__global__ __launch_bounds__(256) void init_state_k(unsigned short* h_bf, float* c, float* prev) {
    int i = blockIdx.x * 256 + threadIdx.x;        // B*H total
    h_bf[i] = 0; c[i] = 0.f;
    if (i < B_) prev[i] = 0.f;
}

// ---------------------------------------------------------------------------
// WMMA GEMM: OUT[M,N] = A[M,K]@W[N,K]^T (+ A2@W2^T) (+ Xf[M,Kf]@Wf[N,Kf]^T)
// (+ bias), optional sigmoid, fp32 and/or bf16 output.
// A/W are bf16; one wave computes a 16x64 tile (4 accumulators), fragments
// built per the CDNA5 ISA VGPR layouts for v_wmma_f32_16x16x32_bf16.
// ---------------------------------------------------------------------------
__global__ __launch_bounds__(256) void gemm_wmma_k(
        const unsigned short* __restrict__ A, const unsigned short* __restrict__ W,
        int N, int K,
        const unsigned short* __restrict__ A2, const unsigned short* __restrict__ W2, int K2,
        const float* __restrict__ Xf, int xfStride,
        const float* __restrict__ Wf, int wfStride, int Kf,
        const float* __restrict__ bias,
        float* __restrict__ outF, unsigned short* __restrict__ outBf, int act) {
    const int lane = threadIdx.x & 31;
    const int wave = threadIdx.x >> 5;
    const int nStrips = N >> 6;
    const int tile = blockIdx.x * (blockDim.x >> 5) + wave;
    const int rowBase = (tile / nStrips) << 4;
    const int colBase = (tile % nStrips) << 6;
    const int half = lane >> 4;                    // lanes 0-15 vs 16-31
    const int mn = lane & 15;

    v8f acc[4] = {};

    const unsigned short* Ap = A;
    const unsigned short* Wp = W;
    int Kc = K;
    for (int pass = 0; pass < 2; ++pass) {
        if (pass == 1) { if (A2 == nullptr) break; Ap = A2; Wp = W2; Kc = K2; }
        for (int k0 = 0; k0 < Kc; k0 += 32) {
            // A fragment: lane holds row rowBase+mn; elems 0..7 = K k0+half*8..,
            // elems 8..15 = K k0+16+half*8.. (ISA 16-bit A 16x32 layout)
            Frag a;
            const u32x4* pa = (const u32x4*)(Ap + (size_t)(rowBase + mn) * Kc + k0 + half * 8);
            a.u[0] = pa[0];
            a.u[1] = pa[2];
#pragma unroll
            for (int t = 0; t < 4; ++t) {
                // B fragment: lane holds col colBase+t*16+mn; elems 0..7 = K k0+half*8..,
                // elems 8..15 = K k0+16+half*8.. (same K order as A).
                Frag b;
                const u32x4* pb = (const u32x4*)(Wp + (size_t)(colBase + t * 16 + mn) * Kc + k0 + half * 8);
                b.u[0] = pb[0];
                b.u[1] = pb[2];
                acc[t] = __builtin_amdgcn_wmma_f32_16x16x32_bf16(
                    false, a.v, false, b.v, (short)0, acc[t], false, false);
            }
        }
    }

    // fp32 scalar tail for the small-K part (input features / inp columns).
    if (Xf != nullptr) {
#pragma unroll
        for (int t = 0; t < 4; ++t) {
            const int col = colBase + t * 16 + mn;
            float wv[8];
            for (int f = 0; f < Kf; ++f) wv[f] = Wf[(size_t)col * wfStride + f];
#pragma unroll
            for (int r = 0; r < 8; ++r) {
                const int row = rowBase + half * 8 + r;
                float s = 0.f;
                for (int f = 0; f < Kf; ++f) s += Xf[(size_t)row * xfStride + f] * wv[f];
                acc[t][r] += s;
            }
        }
    }

    // C/D layout: lane (half,mn): VGPR r <-> row rowBase + half*8 + r, col = mn.
#pragma unroll
    for (int t = 0; t < 4; ++t) {
        const int col = colBase + t * 16 + mn;
        const float bb = bias ? bias[col] : 0.f;
#pragma unroll
        for (int r = 0; r < 8; ++r) {
            const int row = rowBase + half * 8 + r;
            float v = acc[t][r] + bb;
            if (act == 1) v = 1.f / (1.f + expf(-v));
            if (outF)  outF[(size_t)row * N + col] = v;
            if (outBf) outBf[(size_t)row * N + col] = f2bf(v);
        }
    }
}

// ---------------------------------------------------------------------------
// LSTM pointwise cell (gates pre-biased, torch order i,f,g,o).
// ---------------------------------------------------------------------------
__global__ __launch_bounds__(256) void lstm_cell_k(const float* __restrict__ gates,
        float* __restrict__ c, unsigned short* __restrict__ h_bf,
        unsigned short* __restrict__ encOut, int t) {
    int idx = blockIdx.x * 256 + threadIdx.x;      // B*H total
    int b = idx >> 9, j = idx & 511;
    size_t base = (size_t)b * (4 * H_);
    float i_ = sigf(gates[base + j]);
    float f_ = sigf(gates[base + H_ + j]);
    float g_ = tanhf(gates[base + 2 * H_ + j]);
    float o_ = sigf(gates[base + 3 * H_ + j]);
    float cn = f_ * c[idx] + i_ * g_;
    c[idx] = cn;
    float hn = o_ * tanhf(cn);
    unsigned short hb = f2bf(hn);
    h_bf[idx] = hb;
    if (encOut) encOut[(size_t)b * (S_ * H_) + (size_t)t * H_ + j] = hb;
}

__global__ __launch_bounds__(256) void make_inp_k(const float* __restrict__ prev,
        const float* __restrict__ x2, float* __restrict__ inp, int t) {
    int b = blockIdx.x * 256 + threadIdx.x;
    if (b < B_) { inp[2 * b] = prev[b]; inp[2 * b + 1] = x2[(size_t)b * T_ + t]; }
}

__global__ __launch_bounds__(256) void softmax_k(float* __restrict__ scores) {
    __shared__ float red[256];
    const int row = blockIdx.x, tid = threadIdx.x;
    float x = scores[(size_t)row * S_ + tid];
    red[tid] = x; __syncthreads();
    for (int st = 128; st > 0; st >>= 1) {
        if (tid < st) red[tid] = fmaxf(red[tid], red[tid + st]);
        __syncthreads();
    }
    float mx = red[0]; __syncthreads();
    float e = expf(x - mx);
    red[tid] = e; __syncthreads();
    for (int st = 128; st > 0; st >>= 1) {
        if (tid < st) red[tid] += red[tid + st];
        __syncthreads();
    }
    scores[(size_t)row * S_ + tid] = e / red[0];
}

// ctx[b,h] = sum_s w[b,s] * enc[b,s,h]  (batched GEMV; bandwidth bound)
__global__ __launch_bounds__(256) void context_k(const float* __restrict__ w,
        const unsigned short* __restrict__ enc, float* __restrict__ ctx) {
    __shared__ float wv[S_];
    const int b = blockIdx.x, tid = threadIdx.x;
    wv[tid] = w[(size_t)b * S_ + tid];
    __syncthreads();
    const unsigned short* eb = enc + (size_t)b * (S_ * H_);
    float a0 = 0.f, a1 = 0.f;
    for (int s = 0; s < S_; ++s) {
        float ws_ = wv[s];
        a0 += ws_ * bf2f(eb[(size_t)s * H_ + tid]);
        a1 += ws_ * bf2f(eb[(size_t)s * H_ + tid + 256]);
    }
    ctx[(size_t)b * H_ + tid] = a0;
    ctx[(size_t)b * H_ + tid + 256] = a1;
}

// Decoder BatchNorm1d (training-mode batch stats over B) for [inp(2) | ctx(512)].
__global__ __launch_bounds__(256) void bn_d_stats_k(const float* __restrict__ inp,
        const float* __restrict__ ctx, const float* __restrict__ g,
        const float* __restrict__ bb, float* __restrict__ dscale, float* __restrict__ dshift) {
    __shared__ float s1[256], s2[256];
    const int c = blockIdx.x;                      // 0..513
    float sum = 0.f, sq = 0.f;
    for (int b = threadIdx.x; b < B_; b += 256) {
        float v = (c < 2) ? inp[2 * b + c] : ctx[(size_t)b * H_ + (c - 2)];
        sum += v; sq += v * v;
    }
    s1[threadIdx.x] = sum; s2[threadIdx.x] = sq; __syncthreads();
    for (int st = 128; st > 0; st >>= 1) {
        if (threadIdx.x < st) { s1[threadIdx.x] += s1[threadIdx.x + st]; s2[threadIdx.x] += s2[threadIdx.x + st]; }
        __syncthreads();
    }
    if (threadIdx.x == 0) {
        float m = s1[0] / (float)B_;
        float var = s2[0] / (float)B_ - m * m;
        float sc = g[c] * rsqrtf(var + EPS_);
        dscale[c] = sc; dshift[c] = bb[c] - m * sc;
    }
}

__global__ __launch_bounds__(256) void bn_d_apply_k(const float* __restrict__ inp,
        const float* __restrict__ ctx, const float* __restrict__ dscale,
        const float* __restrict__ dshift, unsigned short* __restrict__ apin,
        float* __restrict__ api_inp) {
    const int b = blockIdx.x, tid = threadIdx.x;
    for (int h = tid; h < H_; h += 256)
        apin[(size_t)b * H_ + h] = f2bf(ctx[(size_t)b * H_ + h] * dscale[2 + h] + dshift[2 + h]);
    if (tid < 2) api_inp[2 * b + tid] = inp[2 * b + tid] * dscale[tid] + dshift[tid];
}

// y[b] = tanh(h[b,:]) . out_W + out_b ; writes output column t and prev.
__global__ __launch_bounds__(256) void y_out_k(const unsigned short* __restrict__ h_bf,
        const float* __restrict__ outW, const float* __restrict__ outb,
        float* __restrict__ prev, float* __restrict__ out, int t) {
    __shared__ float red[256];
    const int b = blockIdx.x, tid = threadIdx.x;
    float p = tanhf(bf2f(h_bf[(size_t)b * H_ + tid])) * outW[tid]
            + tanhf(bf2f(h_bf[(size_t)b * H_ + tid + 256])) * outW[tid + 256];
    red[tid] = p; __syncthreads();
    for (int st = 128; st > 0; st >>= 1) {
        if (tid < st) red[tid] += red[tid + st];
        __syncthreads();
    }
    if (tid == 0) {
        float y = red[0] + outb[0];
        out[(size_t)b * T_ + t] = y;
        prev[b] = y;
    }
}

// ---------------------------------------------------------------------------
extern "C" void kernel_launch(void* const* d_in, const int* in_sizes, int n_in,
                              void* d_out, int out_size, void* d_ws, size_t ws_size,
                              hipStream_t stream) {
    (void)in_sizes; (void)n_in; (void)out_size; (void)ws_size;
    const float* x1     = (const float*)d_in[0];
    const float* x2     = (const float*)d_in[1];
    const float* bn_e_g = (const float*)d_in[2];
    const float* bn_e_b = (const float*)d_in[3];
    const float* Wih_e  = (const float*)d_in[4];
    const float* Whh_e  = (const float*)d_in[5];
    const float* bih_e  = (const float*)d_in[6];
    const float* bhh_e  = (const float*)d_in[7];
    const float* attn_W = (const float*)d_in[8];
    const float* attn_b = (const float*)d_in[9];
    const float* comb_W = (const float*)d_in[10];
    const float* comb_b = (const float*)d_in[11];
    const float* bn_d_g = (const float*)d_in[12];
    const float* bn_d_b = (const float*)d_in[13];
    const float* Wih_d  = (const float*)d_in[14];
    const float* Whh_d  = (const float*)d_in[15];
    const float* bih_d  = (const float*)d_in[16];
    const float* bhh_d  = (const float*)d_in[17];
    const float* out_W  = (const float*)d_in[18];
    const float* out_b  = (const float*)d_in[19];

    char* wsb = (char*)d_ws;
    size_t off = 0;
    auto alloc = [&](size_t bytes) -> char* {
        char* p = wsb + off;
        off = (off + bytes + 255) & ~(size_t)255;
        return p;
    };

    float* scaleE  = (float*)alloc(F_ * 4);
    float* shiftE  = (float*)alloc(F_ * 4);
    float* bsum_e  = (float*)alloc(4 * H_ * 4);
    float* bsum_d  = (float*)alloc(4 * H_ * 4);
    unsigned short* WhhE_bf  = (unsigned short*)alloc((size_t)4 * H_ * H_ * 2);
    unsigned short* WihD_bf  = (unsigned short*)alloc((size_t)4 * H_ * H_ * 2);
    unsigned short* WhhD_bf  = (unsigned short*)alloc((size_t)4 * H_ * H_ * 2);
    unsigned short* attnWh_bf = (unsigned short*)alloc((size_t)S_ * H_ * 2);
    unsigned short* combWc_bf = (unsigned short*)alloc((size_t)H_ * H_ * 2);
    float* xn      = (float*)alloc((size_t)S_ * B_ * F_ * 4);
    unsigned short* h_bf = (unsigned short*)alloc((size_t)B_ * H_ * 2);
    float* cbuf    = (float*)alloc((size_t)B_ * H_ * 4);
    float* gates   = (float*)alloc((size_t)B_ * 4 * H_ * 4);
    unsigned short* enc_bf = (unsigned short*)alloc((size_t)B_ * S_ * H_ * 2);   // 128 MB
    float* scores  = (float*)alloc((size_t)B_ * S_ * 4);
    float* ctx     = (float*)alloc((size_t)B_ * H_ * 4);
    unsigned short* apin_bf  = (unsigned short*)alloc((size_t)B_ * H_ * 2);
    float* api_inp = (float*)alloc((size_t)B_ * 2 * 4);
    float* inp     = (float*)alloc((size_t)B_ * 2 * 4);
    unsigned short* xcomb_bf = (unsigned short*)alloc((size_t)B_ * H_ * 2);
    float* dscale  = (float*)alloc((2 + H_) * 4);
    float* dshift  = (float*)alloc((2 + H_) * 4);
    float* prev    = (float*)alloc(B_ * 4);

    // ---- prep ----
    bn_e_stats_k<<<F_, 256, 0, stream>>>(x1, bn_e_g, bn_e_b, scaleE, shiftE);
    build_xn_k<<<(S_ * B_ * F_) / 256, 256, 0, stream>>>(x1, scaleE, shiftE, xn);
    cvt_bf16_k<<<(4 * H_ * H_) / 256, 256, 0, stream>>>(Whh_e, WhhE_bf, H_, H_, 0, 4 * H_ * H_);
    cvt_bf16_k<<<(4 * H_ * H_) / 256, 256, 0, stream>>>(Wih_d, WihD_bf, H_, H_, 0, 4 * H_ * H_);
    cvt_bf16_k<<<(4 * H_ * H_) / 256, 256, 0, stream>>>(Whh_d, WhhD_bf, H_, H_, 0, 4 * H_ * H_);
    cvt_bf16_k<<<(S_ * H_) / 256, 256, 0, stream>>>(attn_W, attnWh_bf, H_, 2 + H_, 2, S_ * H_);
    cvt_bf16_k<<<(H_ * H_) / 256, 256, 0, stream>>>(comb_W, combWc_bf, H_, 2 + H_, 2, H_ * H_);
    vec_add_k<<<(4 * H_) / 256, 256, 0, stream>>>(bih_e, bhh_e, bsum_e, 4 * H_);
    vec_add_k<<<(4 * H_) / 256, 256, 0, stream>>>(bih_d, bhh_d, bsum_d, 4 * H_);
    init_state_k<<<(B_ * H_) / 256, 256, 0, stream>>>(h_bf, cbuf, prev);

    // ---- encoder scan: gates = h@Whh_e^T (WMMA) + xn_t@Wih_e^T (scalar K=8) + bias ----
    for (int t = 0; t < S_; ++t) {
        gemm_wmma_k<<<128, 256, 0, stream>>>(h_bf, WhhE_bf, 4 * H_, H_,
            nullptr, nullptr, 0,
            xn + (size_t)t * B_ * F_, F_, Wih_e, F_, F_,
            bsum_e, gates, nullptr, 0);
        lstm_cell_k<<<(B_ * H_) / 256, 256, 0, stream>>>(gates, cbuf, h_bf, enc_bf, t);
    }

    // ---- decoder scan (h_bf/cbuf continue from encoder final state) ----
    for (int t = 0; t < T_; ++t) {
        make_inp_k<<<2, 256, 0, stream>>>(prev, x2, inp, t);
        // attention scores: [B,514]@[514,S] -> WMMA K=512 + scalar 2 cols + bias
        gemm_wmma_k<<<16, 256, 0, stream>>>(h_bf, attnWh_bf, S_, H_,
            nullptr, nullptr, 0,
            inp, 2, attn_W, 2 + H_, 2,
            attn_b, scores, nullptr, 0);
        softmax_k<<<B_, 256, 0, stream>>>(scores);
        context_k<<<B_, 256, 0, stream>>>(scores, enc_bf, ctx);
        bn_d_stats_k<<<2 + H_, 256, 0, stream>>>(inp, ctx, bn_d_g, bn_d_b, dscale, dshift);
        bn_d_apply_k<<<B_, 256, 0, stream>>>(inp, ctx, dscale, dshift, apin_bf, api_inp);
        // xcomb = sigmoid(api@comb_W^T + b): WMMA K=512 + scalar 2 cols, bf16 out
        gemm_wmma_k<<<32, 256, 0, stream>>>(apin_bf, combWc_bf, H_, H_,
            nullptr, nullptr, 0,
            api_inp, 2, comb_W, 2 + H_, 2,
            comb_b, nullptr, xcomb_bf, 1);
        // decoder gates: xcomb@Wih_d^T + h@Whh_d^T + bias (two WMMA K passes)
        gemm_wmma_k<<<128, 256, 0, stream>>>(xcomb_bf, WihD_bf, 4 * H_, H_,
            h_bf, WhhD_bf, H_,
            nullptr, 0, nullptr, 0, 0,
            bsum_d, gates, nullptr, 0);
        lstm_cell_k<<<(B_ * H_) / 256, 256, 0, stream>>>(gates, cbuf, h_bf, nullptr, 0);
        y_out_k<<<B_, 256, 0, stream>>>(h_bf, out_W, out_b, prev, (float*)d_out, t);
    }
}
